// BCA_58368605552817
// MI455X (gfx1250) — matrix operand order repaired and checked by reading, not compiled, not run on hardware
//
#include <hip/hip_runtime.h>

typedef __attribute__((ext_vector_type(16))) _Float16 v16h;
typedef __attribute__((ext_vector_type(8)))  float    v8f;

constexpr int kH    = 512;
constexpr int kNH   = 8;
constexpr int kHD   = 64;
constexpr int kL    = 1024;
constexpr int kB    = 8;
constexpr int kM    = kB * kL;   // 8192 rows

union FragH {
    v16h v;
    uint4 q[2];
    _Float16 h[16];
};

// WGP-scope prefetch (scope 0): pulls into all cache levels incl. WGP cache.
// __builtin_prefetch only reaches SE/SYS scopes, which stop at GL2.
__device__ __forceinline__ void prefetch_wgp(const void* p) {
    asm volatile("global_prefetch_b8 %0, off" :: "v"(p) : "memory");
}

// ---- fragment loaders (contiguous k order, same for A and B) -----------------------

// A fragment 16x32 (f16): lanes 0-15 -> M=lane, k-base 0; lanes 16-31 -> M=lane-16, k-base 16.
// halves 0..15 = K kb..kb+15
__device__ __forceinline__ FragH load_frag_a(const _Float16* __restrict__ base, int ld) {
    int lane = threadIdx.x & 31;
    int m    = lane & 15;
    int kb   = (lane >> 4) << 4;
    const _Float16* p = base + (size_t)m * ld + kb;
    FragH f;
    f.q[0] = *reinterpret_cast<const uint4*>(p);
    f.q[1] = *reinterpret_cast<const uint4*>(p + 8);
    return f;
}

// B fragment 32x16 (f16) from row-major W[N][K]: lane n=lane&15 holds column n,
// contiguous 16 k-halves at k-base 0 (lanes 0-15) / 16 (lanes 16-31).
__device__ __forceinline__ FragH load_frag_b(const _Float16* __restrict__ base, int ld) {
    int lane = threadIdx.x & 31;
    int n    = lane & 15;
    int kb   = (lane >> 4) << 4;
    const _Float16* p = base + (size_t)n * ld + kb;
    FragH f;
    f.q[0] = *reinterpret_cast<const uint4*>(p);
    f.q[1] = *reinterpret_cast<const uint4*>(p + 8);
    return f;
}

__device__ __forceinline__ v8f wmma16(FragH a, FragH b, v8f c) {
    return __builtin_amdgcn_wmma_f32_16x16x32_f16(false, a.v, false, b.v,
                                                  (short)0, c, false, false);
}

// ---- f32 -> f16 conversion (with scale, used for 0.5*(ctx+ctx)) ------------

__global__ void cvt_f32_f16(const float* __restrict__ s, _Float16* __restrict__ d,
                            float scale, int n) {
    int i = blockIdx.x * blockDim.x + threadIdx.x;
    if (i < n) d[i] = (_Float16)(s[i] * scale);
}

// ---- WMMA GEMM: Y[m][n] = sum_k X[m][k] * W[n][k] + bias[n] ----------------
// X row-major [M][ldx] f16, W row-major [N][ldw] f16 (computes X @ W.T).
// Block = 256 threads = 8 waves in 4x2; block tile 128x128; wave tile 32x64.
// Explicit 1-deep software pipeline over 64-wide k-steps (K % 64 == 0).
__global__ __launch_bounds__(256) void gemm_wmma(
    const _Float16* __restrict__ X, int ldx,
    const _Float16* __restrict__ W, int ldw,
    const float* __restrict__ bias,
    _Float16* __restrict__ Yh, float* __restrict__ Yf, int ldy,
    int K)
{
    int wave = threadIdx.x >> 5;
    int wm = wave >> 1, wn = wave & 1;
    int bm = blockIdx.x * 128 + wm * 32;
    int bn = blockIdx.y * 128 + wn * 64;
    int lane = threadIdx.x & 31;

    v8f acc[2][4] = {};
    const _Float16* xa = X + (size_t)bm * ldx;
    const _Float16* wb = W + (size_t)bn * ldw;

    FragH a0A, a1A, bA0, bA1, bA2, bA3;
    FragH a0B, a1B, bB0, bB1, bB2, bB3;

    auto loadA = [&](int k) {
        a0A = load_frag_a(xa + k, ldx);
        a1A = load_frag_a(xa + (size_t)16 * ldx + k, ldx);
        bA0 = load_frag_b(wb + k, ldw);
        bA1 = load_frag_b(wb + (size_t)16 * ldw + k, ldw);
        bA2 = load_frag_b(wb + (size_t)32 * ldw + k, ldw);
        bA3 = load_frag_b(wb + (size_t)48 * ldw + k, ldw);
    };
    auto loadB = [&](int k) {
        a0B = load_frag_a(xa + k, ldx);
        a1B = load_frag_a(xa + (size_t)16 * ldx + k, ldx);
        bB0 = load_frag_b(wb + k, ldw);
        bB1 = load_frag_b(wb + (size_t)16 * ldw + k, ldw);
        bB2 = load_frag_b(wb + (size_t)32 * ldw + k, ldw);
        bB3 = load_frag_b(wb + (size_t)48 * ldw + k, ldw);
    };

    loadA(0);
    for (int k0 = 0; k0 < K; k0 += 64) {
        // WGP-scope prefetch ~1.5 steps ahead (clamped in-range)
        int kpf = (k0 + 96 < K) ? (k0 + 96) : 0;
        prefetch_wgp(xa + (size_t)lane * ldx + kpf);
        prefetch_wgp(wb + (size_t)lane * ldw + kpf);
        prefetch_wgp(wb + (size_t)(32 + lane) * ldw + kpf);

        loadB(k0 + 32);                // stage next half-step
        acc[0][0] = wmma16(a0A, bA0, acc[0][0]);
        acc[1][0] = wmma16(a1A, bA0, acc[1][0]);
        acc[0][1] = wmma16(a0A, bA1, acc[0][1]);
        acc[1][1] = wmma16(a1A, bA1, acc[1][1]);
        acc[0][2] = wmma16(a0A, bA2, acc[0][2]);
        acc[1][2] = wmma16(a1A, bA2, acc[1][2]);
        acc[0][3] = wmma16(a0A, bA3, acc[0][3]);
        acc[1][3] = wmma16(a1A, bA3, acc[1][3]);

        int kn = (k0 + 64 < K) ? (k0 + 64) : 0;   // last iter: dummy reload
        loadA(kn);
        acc[0][0] = wmma16(a0B, bB0, acc[0][0]);
        acc[1][0] = wmma16(a1B, bB0, acc[1][0]);
        acc[0][1] = wmma16(a0B, bB1, acc[0][1]);
        acc[1][1] = wmma16(a1B, bB1, acc[1][1]);
        acc[0][2] = wmma16(a0B, bB2, acc[0][2]);
        acc[1][2] = wmma16(a1B, bB2, acc[1][2]);
        acc[0][3] = wmma16(a0B, bB3, acc[0][3]);
        acc[1][3] = wmma16(a1B, bB3, acc[1][3]);
    }

    int ncol = lane & 15;
    int rhi  = (lane >> 4) << 3;
#pragma unroll
    for (int j = 0; j < 4; j++) {
        float bj = bias ? bias[bn + j * 16 + ncol] : 0.f;
#pragma unroll
        for (int i = 0; i < 2; i++) {
#pragma unroll
            for (int v = 0; v < 8; v++) {
                int row = bm + i * 16 + rhi + v;
                float val = acc[i][j][v] + bj;
                size_t idx = (size_t)row * ldy + bn + j * 16 + ncol;
                if (Yf) Yf[idx] = val;
                else    Yh[idx] = (_Float16)val;
            }
        }
    }
}

// ---- Flash attention with WMMA ---------------------------------------------
// grid = (L/128, NH, B); block = 256 (8 waves, each owns 16 query rows).
// 64-key blocks: K staged in LDS (shared by all waves), V staged transposed.
// Softmax scale 1/sqrt(64) and log2(e) folded into Q; exp2f -> v_exp_f32.
__global__ __launch_bounds__(256) void attn_wmma(
    const _Float16* __restrict__ Qg, int ldq,
    const _Float16* __restrict__ Kg, int ldk,
    const _Float16* __restrict__ Vg, int ldv,
    float* __restrict__ Out, int ldo,
    int accumulate)
{
    constexpr int KP = 72;                    // LDS pitch (halves): 144B rows
    __shared__ _Float16 kls[64][KP];          // K tile  [key][d]
    __shared__ _Float16 vt [64][KP];          // V^T tile [d][key]
    __shared__ _Float16 pls[8][16 * KP];      // per-wave P tile 16x64

    int b = blockIdx.z, h = blockIdx.y;
    int wave = threadIdx.x >> 5;
    int lane = threadIdx.x & 31;
    int qrow0 = blockIdx.x * 128 + wave * 16;

    const _Float16* Qb = Qg + ((size_t)b * kL) * ldq + h * kHD;
    const _Float16* Kb = Kg + ((size_t)b * kL) * ldk + h * kHD;
    const _Float16* Vb = Vg + ((size_t)b * kL) * ldv + h * kHD;

    FragH qf0 = load_frag_a(Qb + (size_t)qrow0 * ldq, ldq);       // d 0..31
    FragH qf1 = load_frag_a(Qb + (size_t)qrow0 * ldq + 32, ldq);  // d 32..63
    const float qs = 0.125f * 1.44269504f;    // 1/sqrt(HD) * log2(e)
#pragma unroll
    for (int i = 0; i < 16; i++) {
        qf0.h[i] = (_Float16)((float)qf0.h[i] * qs);
        qf1.h[i] = (_Float16)((float)qf1.h[i] * qs);
    }

    v8f ctx[4] = {};
    float rmax[8], rsum[8];
#pragma unroll
    for (int v = 0; v < 8; v++) { rmax[v] = -1e30f; rsum[v] = 0.f; }

    int ncol = lane & 15;
    _Float16* pw = &pls[wave][0];

    int skey = threadIdx.x >> 2;              // 0..63  (staging row)
    int sx   = (threadIdx.x & 3) << 4;        // 0,16,32,48 (staging col)

    for (int k0 = 0; k0 < kL; k0 += 64) {
        // ---- cooperative stage: K (row-major) + V^T, WGP prefetch next ----
        __syncthreads();
        {
            const _Float16* ksrc = Kb + (size_t)(k0 + skey) * ldk + sx;
            const _Float16* vsrc = Vb + (size_t)(k0 + skey) * ldv + sx;
            uint4 kr0 = *reinterpret_cast<const uint4*>(ksrc);
            uint4 kr1 = *reinterpret_cast<const uint4*>(ksrc + 8);
            uint4 vr0 = *reinterpret_cast<const uint4*>(vsrc);
            uint4 vr1 = *reinterpret_cast<const uint4*>(vsrc + 8);
            size_t pofs = (size_t)((k0 + 64 < kL) ? 64 : 0);
            prefetch_wgp(ksrc + pofs * ldk);
            prefetch_wgp(vsrc + pofs * ldv);
            *reinterpret_cast<uint4*>(&kls[skey][sx])     = kr0;
            *reinterpret_cast<uint4*>(&kls[skey][sx + 8]) = kr1;
            _Float16 tmp[16];
            *reinterpret_cast<uint4*>(&tmp[0]) = vr0;
            *reinterpret_cast<uint4*>(&tmp[8]) = vr1;
#pragma unroll
            for (int i = 0; i < 16; i++) vt[sx + i][skey] = tmp[i];
        }
        __syncthreads();

        // ---- scores: 4 key tiles of 16, contraction over d in two 32-slices
        v8f s[4] = {};
        {
            int kb = (lane >> 4) << 4;
#pragma unroll
            for (int j = 0; j < 4; j++) {
                const _Float16* p0 = &kls[j * 16 + ncol][kb];
                const _Float16* p1 = &kls[j * 16 + ncol][32 + kb];
                FragH b0, b1;
                b0.q[0] = *reinterpret_cast<const uint4*>(p0);
                b0.q[1] = *reinterpret_cast<const uint4*>(p0 + 8);
                b1.q[0] = *reinterpret_cast<const uint4*>(p1);
                b1.q[1] = *reinterpret_cast<const uint4*>(p1 + 8);
                s[j] = wmma16(qf0, b0, s[j]);
                s[j] = wmma16(qf1, b1, s[j]);
            }
        }

        // ---- online softmax over 64 keys (log2 domain) ----
#pragma unroll
        for (int v = 0; v < 8; v++) {
            float a0 = s[0][v], a1 = s[1][v], a2 = s[2][v], a3 = s[3][v];
            float mx = fmaxf(fmaxf(a0, a1), fmaxf(a2, a3));
#pragma unroll
            for (int off = 1; off < 16; off <<= 1)
                mx = fmaxf(mx, __shfl_xor(mx, off, 32));
            float mnew = fmaxf(rmax[v], mx);
            float sc = exp2f(rmax[v] - mnew);
            rmax[v] = mnew;
            float p0 = exp2f(a0 - mnew);
            float p1 = exp2f(a1 - mnew);
            float p2 = exp2f(a2 - mnew);
            float p3 = exp2f(a3 - mnew);
            float ps = (p0 + p1) + (p2 + p3);
#pragma unroll
            for (int off = 1; off < 16; off <<= 1)
                ps += __shfl_xor(ps, off, 32);
            rsum[v] = rsum[v] * sc + ps;
            ctx[0][v] *= sc; ctx[1][v] *= sc; ctx[2][v] *= sc; ctx[3][v] *= sc;
            int rr = ((lane >> 4) << 3) + v;
            _Float16* pr = pw + rr * KP + ncol;
            pr[0]  = (_Float16)p0;
            pr[16] = (_Float16)p1;
            pr[32] = (_Float16)p2;
            pr[48] = (_Float16)p3;
        }
        __asm__ volatile("s_wait_dscnt 0" ::: "memory");

        // ---- P fragments (keys 0..31 / 32..63) from LDS ----
        FragH pa0, pa1;
        {
            int m  = lane & 15;
            int kb = (lane >> 4) << 4;
            const _Float16* p = pw + m * KP + kb;
            pa0.q[0] = *reinterpret_cast<const uint4*>(p);
            pa0.q[1] = *reinterpret_cast<const uint4*>(p + 8);
            pa1.q[0] = *reinterpret_cast<const uint4*>(p + 32);
            pa1.q[1] = *reinterpret_cast<const uint4*>(p + 40);
        }
        // ---- ctx += P @ V (B fragments from V^T) ----
        {
            int kb2 = (lane >> 4) << 4;
#pragma unroll
            for (int t2 = 0; t2 < 4; t2++) {
                const _Float16* p0 = &vt[t2 * 16 + ncol][kb2];
                const _Float16* p1 = &vt[t2 * 16 + ncol][32 + kb2];
                FragH vb0, vb1;
                vb0.q[0] = *reinterpret_cast<const uint4*>(p0);
                vb0.q[1] = *reinterpret_cast<const uint4*>(p0 + 8);
                vb1.q[0] = *reinterpret_cast<const uint4*>(p1);
                vb1.q[1] = *reinterpret_cast<const uint4*>(p1 + 8);
                ctx[t2] = wmma16(pa0, vb0, ctx[t2]);
                ctx[t2] = wmma16(pa1, vb1, ctx[t2]);
            }
        }
    }

    // ---- epilogue: normalize + (optionally accumulating) f32 store ----
    int rhi = (lane >> 4) << 3;
#pragma unroll
    for (int v = 0; v < 8; v++) {
        float inv = 1.f / rsum[v];
        int row = qrow0 + rhi + v;
        float* o = Out + ((size_t)b * kL + row) * ldo + h * kHD + ncol;
#pragma unroll
        for (int t2 = 0; t2 < 4; t2++) {
            float val = ctx[t2][v] * inv;
            if (accumulate) o[t2 * 16] += val;
            else            o[t2 * 16]  = val;
        }
    }
}

// ---------------------------------------------------------------------------

extern "C" void kernel_launch(void* const* d_in, const int* in_sizes, int n_in,
                              void* d_out, int out_size, void* d_ws, size_t ws_size,
                              hipStream_t stream) {
    const float* hidden    = (const float*)d_in[0];
    const float* text      = (const float*)d_in[1];
    const float* w_q_img   = (const float*)d_in[2];
    const float* b_q_img   = (const float*)d_in[3];
    const float* w_k_img   = (const float*)d_in[4];
    const float* b_k_img   = (const float*)d_in[5];
    const float* w_v_img   = (const float*)d_in[6];
    const float* b_v_img   = (const float*)d_in[7];
    const float* w_q_txt   = (const float*)d_in[8];
    const float* b_q_txt   = (const float*)d_in[9];
    const float* w_k_txt   = (const float*)d_in[10];
    const float* b_k_txt   = (const float*)d_in[11];
    const float* w_v_txt   = (const float*)d_in[12];
    const float* b_v_txt   = (const float*)d_in[13];
    const float* w_out_img = (const float*)d_in[14];
    const float* b_out_img = (const float*)d_in[15];
    const float* w_out_txt = (const float*)d_in[16];
    const float* b_out_txt = (const float*)d_in[17];
    const float* w_cat     = (const float*)d_in[18];
    const float* b_cat     = (const float*)d_in[19];
    const float* in_proj_w = (const float*)d_in[20];
    const float* in_proj_b = (const float*)d_in[21];
    const float* out_proj_w= (const float*)d_in[22];
    const float* out_proj_b= (const float*)d_in[23];

    char* ws = (char*)d_ws;
    size_t off = 0;
    auto alloc = [&](size_t bytes) -> void* {
        void* p = ws + off;
        off += (bytes + 255) & ~(size_t)255;
        return p;
    };

    const size_t ACT16 = (size_t)kM * kH * sizeof(_Float16);   // 8 MB
    const size_t ACT32 = (size_t)kM * kH * sizeof(float);      // 16 MB

    _Float16* hx16  = (_Float16*)alloc(ACT16);
    _Float16* tx16  = (_Float16*)alloc(ACT16);
    _Float16* q_img = (_Float16*)alloc(ACT16);
    _Float16* k_img = (_Float16*)alloc(ACT16);
    _Float16* v_img = (_Float16*)alloc(ACT16);
    _Float16* q_txt = (_Float16*)alloc(ACT16);
    _Float16* k_txt = (_Float16*)alloc(ACT16);
    _Float16* v_txt = (_Float16*)alloc(ACT16);
    float*    ctxA  = (float*)alloc(ACT32);
    float*    ctxB  = (float*)alloc(ACT32);

    _Float16* w16_qi = (_Float16*)alloc(512 * 512 * 2);
    _Float16* w16_ki = (_Float16*)alloc(512 * 512 * 2);
    _Float16* w16_vi = (_Float16*)alloc(512 * 512 * 2);
    _Float16* w16_qt = (_Float16*)alloc(512 * 512 * 2);
    _Float16* w16_kt = (_Float16*)alloc(512 * 512 * 2);
    _Float16* w16_vt = (_Float16*)alloc(512 * 512 * 2);
    _Float16* w16_oi = (_Float16*)alloc(512 * 512 * 2);
    _Float16* w16_ot = (_Float16*)alloc(512 * 512 * 2);
    _Float16* w16_op = (_Float16*)alloc(512 * 512 * 2);
    _Float16* w16_cat= (_Float16*)alloc(512 * 1024 * 2);
    _Float16* w16_in = (_Float16*)alloc(1536 * 512 * 2);

    // aliases over dead buffers (stream-ordered, hazard-checked):
    _Float16* ctxA16 = hx16;                 // hidden f16 dead after projections
    _Float16* ctxB16 = tx16;
    _Float16* Xcat   = q_img;                // 8192x1024 spans q_img+k_img (contiguous)
    _Float16* outcat = v_img;
    _Float16* qkvp   = q_txt;                // 8192x1536 spans q_txt..v_txt
    float*    ctxP   = ctxA;
    _Float16* ctxP16 = (_Float16*)ctxB;

    auto cvt = [&](const float* s, _Float16* dst, float sc, int n) {
        cvt_f32_f16<<<(n + 255) / 256, 256, 0, stream>>>(s, dst, sc, n);
    };
    auto gemm = [&](const _Float16* X, int ldx, const _Float16* W, int ldw,
                    const float* bias, _Float16* Yh, float* Yf, int ldy,
                    int N, int K) {
        dim3 g(kM / 128, N / 128);
        gemm_wmma<<<g, 256, 0, stream>>>(X, ldx, W, ldw, bias, Yh, Yf, ldy, K);
    };
    auto attn = [&](const _Float16* Q, int ldq, const _Float16* K, int ldk,
                    const _Float16* V, int ldv, float* O, int acc) {
        dim3 g(kL / 128, kNH, kB);
        attn_wmma<<<g, 256, 0, stream>>>(Q, ldq, K, ldk, V, ldv, O, kH, acc);
    };

    const int nAct = kM * kH;

    // phase 0: f32 -> f16 conversions
    cvt(hidden, hx16, 1.f, nAct);
    cvt(text,   tx16, 1.f, nAct);
    cvt(w_q_img, w16_qi, 1.f, 512 * 512);
    cvt(w_k_img, w16_ki, 1.f, 512 * 512);
    cvt(w_v_img, w16_vi, 1.f, 512 * 512);
    cvt(w_q_txt, w16_qt, 1.f, 512 * 512);
    cvt(w_k_txt, w16_kt, 1.f, 512 * 512);
    cvt(w_v_txt, w16_vt, 1.f, 512 * 512);
    cvt(w_out_img, w16_oi, 1.f, 512 * 512);
    cvt(w_out_txt, w16_ot, 1.f, 512 * 512);
    cvt(out_proj_w, w16_op, 1.f, 512 * 512);
    cvt(w_cat, w16_cat, 1.f, 512 * 1024);
    cvt(in_proj_w, w16_in, 1.f, 1536 * 512);

    // phase 1: QKV projections
    gemm(hx16, kH, w16_qi, kH, b_q_img, q_img, nullptr, kH, 512, 512);
    gemm(hx16, kH, w16_ki, kH, b_k_img, k_img, nullptr, kH, 512, 512);
    gemm(hx16, kH, w16_vi, kH, b_v_img, v_img, nullptr, kH, 512, 512);
    gemm(tx16, kH, w16_qt, kH, b_q_txt, q_txt, nullptr, kH, 512, 512);
    gemm(tx16, kH, w16_kt, kH, b_k_txt, k_txt, nullptr, kH, 512, 512);
    gemm(tx16, kH, w16_vt, kH, b_v_txt, v_txt, nullptr, kH, 512, 512);

    // phase 2: four attentions, fused pairs accumulate in f32
    attn(q_img, kH, k_img, kH, v_img, kH, ctxA, 0);   // ctx_img
    attn(q_img, kH, k_txt, kH, v_txt, kH, ctxA, 1);   // + ctx_it
    attn(q_txt, kH, k_txt, kH, v_txt, kH, ctxB, 0);   // ctx_txt
    attn(q_txt, kH, k_img, kH, v_img, kH, ctxB, 1);   // + ctx_ti

    // phase 3: 0.5 * (sum) -> f16
    cvt(ctxA, ctxA16, 0.5f, nAct);
    cvt(ctxB, ctxB16, 0.5f, nAct);

    // phase 4: output projections, written into concat buffer [8192 x 1024]
    gemm(ctxA16, kH, w16_oi, kH, b_out_img, Xcat,       nullptr, 1024, 512, 512);
    gemm(ctxB16, kH, w16_ot, kH, b_out_txt, Xcat + 512, nullptr, 1024, 512, 512);

    // phase 5: cat projection (K = 1024)
    gemm(Xcat, 1024, w16_cat, 1024, b_cat, outcat, nullptr, kH, 512, 1024);

    // phase 6: in_proj -> qkv pool [8192 x 1536]
    gemm(outcat, kH, w16_in, kH, in_proj_b, qkvp, nullptr, 1536, 1536, 512);

    // phase 7: pooling self-attention (head-sliced views of qkv pool)
    attn(qkvp, 1536, qkvp + 512, 1536, qkvp + 1024, 1536, ctxP, 0);

    // phase 8: ctx -> f16
    cvt(ctxP, ctxP16, 1.f, nAct);

    // phase 9: final out_proj, f32 output
    gemm(ctxP16, kH, w16_op, kH, out_proj_b, nullptr, (float*)d_out, kH, 512, 512);
}
